// DNCCell_75823352644053
// MI455X (gfx1250) — matrix operand order repaired and checked by reading, not compiled, not run on hardware
//
#include <hip/hip_runtime.h>
#include <hip/hip_bf16.h>

// Problem constants (match reference)
#define B_   64
#define U_   1024
#define H_   16
#define M_   512
#define D_   4
#define HS_  64
#define K_   1024
#define MP1  (M_ + 1)

typedef __attribute__((ext_vector_type(16))) __bf16 v16bf;
typedef __attribute__((ext_vector_type(8)))  float  v8f;

union BfPack4 { __bf16 b[4]; uint2 u; };       // 4 bf16 -> one 8B store
union BfFrag  { uint4 q[2]; v16bf v; };        // two b128 loads -> A fragment

__device__ __forceinline__ float wave_max(float v) {
#pragma unroll
  for (int off = 16; off > 0; off >>= 1) v = fmaxf(v, __shfl_xor(v, off, 32));
  return v;
}
__device__ __forceinline__ float wave_sum(float v) {
#pragma unroll
  for (int off = 16; off > 0; off >>= 1) v += __shfl_xor(v, off, 32);
  return v;
}

__device__ __forceinline__ uint2 pack_bf4(float a, float b, float c, float d) {
  BfPack4 p;
  p.b[0] = (__bf16)a; p.b[1] = (__bf16)b; p.b[2] = (__bf16)c; p.b[3] = (__bf16)d;
  return p.u;
}

// ---------------------------------------------------------------------------
// K0: copy input memory into the output-resident memory buffer, compute
//     column sums, and write memstate (bf16, GEMM A operand) = (colsum+x)/513.
// One thread per (b, u/4); b128 traffic, coalesced 512B per wave.
// ---------------------------------------------------------------------------
__global__ __launch_bounds__(256)
void init_mem_memstate(const float4* __restrict__ memIn,
                       const float4* __restrict__ x,
                       float4* __restrict__ mem,
                       uint2* __restrict__ memstate_bf) {
  int idx = blockIdx.x * blockDim.x + threadIdx.x;     // b*(U/4) + u4
  if (idx >= B_ * (U_ / 4)) return;
  int b  = idx >> 8;
  int u4 = idx & (U_ / 4 - 1);
  const float4* src = memIn + (size_t)b * (M_ * U_ / 4) + u4;
  float4*       dst = mem   + (size_t)b * (M_ * U_ / 4) + u4;
  float4 s = {0.f, 0.f, 0.f, 0.f};
#pragma unroll 4
  for (int m = 0; m < M_; ++m) {
    float4 v = src[(size_t)m * (U_ / 4)];
    dst[(size_t)m * (U_ / 4)] = v;
    s.x += v.x; s.y += v.y; s.z += v.z; s.w += v.w;
  }
  float4 xv = x[idx];
  const float inv = 1.0f / (float)MP1;
  memstate_bf[idx] = pack_bf4((s.x + xv.x) * inv, (s.y + xv.y) * inv,
                              (s.z + xv.z) * inv, (s.w + xv.w) * inv);
}

// ---------------------------------------------------------------------------
// GEMM: C[64,N] = act(A[64,1024](bf16) @ W[1024,N](f32->bf16) + bias[N])
// bf16 WMMA 16x16x32, f32 accumulate. One wave owns ONE column tile and ALL
// FOUR row tiles: each B fragment (16 immediate-offset loads + 8 pk-cvt)
// feeds 4 WMMAs; A fragments are pre-converted bf16, two b128 loads each.
// N compile-time: all element offsets fold into 24-bit immediates.
// ---------------------------------------------------------------------------
template <int RELU, int N, int WRITE_BF16>
__global__ __launch_bounds__(256)
void gemm_bias_wmma(const __bf16* __restrict__ A,   // [64,1024] bf16 row-major
                    const float* __restrict__ W,    // [1024,N]  f32 row-major
                    const float* __restrict__ bias, // [N]
                    float* __restrict__ C,          // [64,N]
                    __bf16* __restrict__ Cbf) {     // optional bf16 copy of C
  constexpr int tilesN = N >> 4;
  const int lane = threadIdx.x & 31;
  const int tile = __builtin_amdgcn_readfirstlane(
      blockIdx.x * (blockDim.x >> 5) + (threadIdx.x >> 5));
  if (tile >= tilesN) return;                  // scalar-uniform branch
  const int khalf = lane >> 4;                 // lane half selects K sub-group
  const int l15   = lane & 15;
  const int n     = tile * 16 + l15;

  // A fragment bases for the 4 row tiles (row = tm*16 + l15)
  const __bf16* pa0 = A + ((size_t)(l15)      * K_) + khalf * 8;
  const __bf16* pa1 = A + ((size_t)(l15 + 16) * K_) + khalf * 8;
  const __bf16* pa2 = A + ((size_t)(l15 + 32) * K_) + khalf * 8;
  const __bf16* pa3 = A + ((size_t)(l15 + 48) * K_) + khalf * 8;
  const float*  pw  = W + (size_t)khalf * 8 * N + n;

  v8f acc0 = {}, acc1 = {}, acc2 = {}, acc3 = {};
#pragma unroll 2
  for (int kb = 0; kb < K_; kb += 32) {
    // ---- B fragment: K = kb + khalf*8 + (e<8 ? e : e+8); immediate offsets
    float wb[16];
#pragma unroll
    for (int e = 0; e < 16; ++e) wb[e] = pw[(size_t)(e < 8 ? e : e + 8) * N];
    v16bf bv;
#pragma unroll
    for (int e = 0; e < 16; ++e) bv[e] = (__bf16)wb[e];

    if (kb + 32 < K_) __builtin_prefetch(pw + (size_t)32 * N, 0, 1);

    // ---- 4 row tiles: A frag = 2 x b128 bf16 loads (K ..+7 and +16..+23)
    BfFrag a0; a0.q[0] = *(const uint4*)(pa0); a0.q[1] = *(const uint4*)(pa0 + 16);
    BfFrag a1; a1.q[0] = *(const uint4*)(pa1); a1.q[1] = *(const uint4*)(pa1 + 16);
    BfFrag a2; a2.q[0] = *(const uint4*)(pa2); a2.q[1] = *(const uint4*)(pa2 + 16);
    BfFrag a3; a3.q[0] = *(const uint4*)(pa3); a3.q[1] = *(const uint4*)(pa3 + 16);

    acc0 = __builtin_amdgcn_wmma_f32_16x16x32_bf16(false, a0.v, false, bv, (short)0, acc0, false, false);
    acc1 = __builtin_amdgcn_wmma_f32_16x16x32_bf16(false, a1.v, false, bv, (short)0, acc1, false, false);
    acc2 = __builtin_amdgcn_wmma_f32_16x16x32_bf16(false, a2.v, false, bv, (short)0, acc2, false, false);
    acc3 = __builtin_amdgcn_wmma_f32_16x16x32_bf16(false, a3.v, false, bv, (short)0, acc3, false, false);

    pa0 += 32; pa1 += 32; pa2 += 32; pa3 += 32;
    pw  += (size_t)32 * N;
  }

  // ---- epilogue: bias + activation + store (C row = tm*16 + khalf*8 + r)
  const float bz = bias[n];
#pragma unroll
  for (int tm = 0; tm < 4; ++tm) {
    v8f acc = (tm == 0) ? acc0 : (tm == 1) ? acc1 : (tm == 2) ? acc2 : acc3;
    const int mb = tm * 16 + khalf * 8;
#pragma unroll
    for (int r = 0; r < 8; ++r) {
      float v = acc[r] + bz;
      if (RELU) v = fmaxf(v, 0.f);
      C[(size_t)(mb + r) * N + n] = v;
      if (WRITE_BF16) Cbf[(size_t)(mb + r) * N + n] = (__bf16)v;
    }
  }
}

// ---------------------------------------------------------------------------
// Softmax over slots: p viewed as [B, L, H], softmax over L for each (b,h).
// One wave per (b,h); lanes stride the slot axis; shuffle reductions (wave32).
// ---------------------------------------------------------------------------
__global__ __launch_bounds__(256)
void softmax_slots(float* __restrict__ p, int L) {
  const int gwave = (blockIdx.x * blockDim.x + threadIdx.x) >> 5;
  const int lane  = threadIdx.x & 31;
  if (gwave >= B_ * H_) return;
  const int b = gwave >> 4;
  const int h = gwave & 15;
  float* row = p + (size_t)b * ((size_t)L * H_) + h;

  float mx = -3.4e38f;
  for (int m = lane; m < L; m += 32) mx = fmaxf(mx, row[(size_t)m * H_]);
  mx = wave_max(mx);

  float s = 0.f;
  for (int m = lane; m < L; m += 32) {
    float e = __expf(row[(size_t)m * H_] - mx);
    row[(size_t)m * H_] = e;
    s += e;
  }
  s = wave_sum(s);
  const float inv = 1.0f / s;
  for (int m = lane; m < L; m += 32) row[(size_t)m * H_] *= inv;
}

// ---------------------------------------------------------------------------
// Weighted read: r[b,u] = sum_m rw[b,m,h(u)] * mem[b,m,u] + rw[b,M,h]*prev[b,u]
// float4 per thread; result written as packed bf16 (GEMM A operand only).
// ---------------------------------------------------------------------------
__global__ __launch_bounds__(256)
void weighted_read(const float4* __restrict__ mem,
                   const float* __restrict__ rw,     // [B, M+1, H]
                   const float4* __restrict__ prevOut,
                   uint2* __restrict__ r_bf) {
  int idx = blockIdx.x * blockDim.x + threadIdx.x;   // b*(U/4) + u4
  if (idx >= B_ * (U_ / 4)) return;
  int b  = idx >> 8;
  int u4 = idx & (U_ / 4 - 1);
  int h  = u4 >> 4;                                  // (u4*4)/HS
  const float4* mp = mem + (size_t)b * (M_ * U_ / 4) + u4;
  const float*  wp = rw  + (size_t)b * (MP1 * H_) + h;
  float4 acc = {0.f, 0.f, 0.f, 0.f};
#pragma unroll 4
  for (int m = 0; m < M_; ++m) {
    float  w = wp[(size_t)m * H_];
    float4 v = mp[(size_t)m * (U_ / 4)];
    acc.x += w * v.x; acc.y += w * v.y; acc.z += w * v.z; acc.w += w * v.w;
  }
  float  wl = wp[(size_t)M_ * H_];
  float4 pv = prevOut[idx];
  r_bf[idx] = pack_bf4(acc.x + wl * pv.x, acc.y + wl * pv.y,
                       acc.z + wl * pv.z, acc.w + wl * pv.w);
}

// ---------------------------------------------------------------------------
// Memory update: mem = (1-w)*mem + w*newval; fused next-layer memstate (bf16)
// or, on the last layer, copy newval into the final output.
// ---------------------------------------------------------------------------
__global__ __launch_bounds__(256)
void mem_update(float4* __restrict__ mem,
                const float* __restrict__ ww,        // [B, M, H]
                const float4* __restrict__ newval,
                uint2* __restrict__ memstate_bf,
                float4* __restrict__ outFinal,
                int isLast) {
  int idx = blockIdx.x * blockDim.x + threadIdx.x;   // b*(U/4) + u4
  if (idx >= B_ * (U_ / 4)) return;
  int b  = idx >> 8;
  int u4 = idx & (U_ / 4 - 1);
  int h  = u4 >> 4;
  const float4 nv = newval[idx];
  float4*       mp = mem + (size_t)b * (M_ * U_ / 4) + u4;
  const float*  wp = ww  + (size_t)b * (M_ * H_) + h;
  float4 s = {0.f, 0.f, 0.f, 0.f};
#pragma unroll 4
  for (int m = 0; m < M_; ++m) {
    float  w = wp[(size_t)m * H_];
    float4 v = mp[(size_t)m * (U_ / 4)];
    v.x += w * (nv.x - v.x);
    v.y += w * (nv.y - v.y);
    v.z += w * (nv.z - v.z);
    v.w += w * (nv.w - v.w);
    mp[(size_t)m * (U_ / 4)] = v;
    s.x += v.x; s.y += v.y; s.z += v.z; s.w += v.w;
  }
  if (isLast) {
    outFinal[idx] = nv;
  } else {
    const float inv = 1.0f / (float)MP1;
    memstate_bf[idx] = pack_bf4((s.x + nv.x) * inv, (s.y + nv.y) * inv,
                                (s.z + nv.z) * inv, (s.w + nv.w) * inv);
  }
}

// ---------------------------------------------------------------------------
extern "C" void kernel_launch(void* const* d_in, const int* in_sizes, int n_in,
                              void* d_out, int out_size, void* d_ws, size_t ws_size,
                              hipStream_t stream) {
  const float* x   = (const float*)d_in[0];
  const float* mi  = (const float*)d_in[1];
  const float* Wr  = (const float*)d_in[2];
  const float* br  = (const float*)d_in[3];
  const float* Ww  = (const float*)d_in[4];
  const float* bw  = (const float*)d_in[5];
  const float* Wk  = (const float*)d_in[6];
  const float* bk  = (const float*)d_in[7];

  float* outv = (float*)d_out;           // [B,U]
  float* mem  = outv + B_ * U_;          // [B, M*U] — memory state lives in d_out

  // Workspace layout (16B-aligned slabs)
  char* wsb = (char*)d_ws;
  __bf16* memstate_bf = (__bf16*)wsb;  wsb += (size_t)B_ * U_ * 2;        // 128 KB
  __bf16* rvec_bf     = (__bf16*)wsb;  wsb += (size_t)B_ * U_ * 2;        // 128 KB
  __bf16* newv_bf     = (__bf16*)wsb;  wsb += (size_t)B_ * U_ * 2;        // 128 KB
  float*  newv        = (float*)wsb;   wsb += (size_t)B_ * U_ * 4;        // 256 KB
  float*  rlog        = (float*)wsb;   wsb += (size_t)B_ * MP1 * H_ * 4;  // ~2.1 MB
  float*  wlog        = (float*)wsb;   wsb += (size_t)B_ * M_ * H_ * 4;   // 2 MB

  const int TPB = 256;
  const int nBU4 = B_ * U_ / 4;
  dim3 gBU4((nBU4 + TPB - 1) / TPB);

  init_mem_memstate<<<gBU4, TPB, 0, stream>>>(
      (const float4*)mi, (const float4*)x, (float4*)mem, (uint2*)memstate_bf);

  constexpr int NR = H_ * MP1;   // 8208
  constexpr int NW = H_ * M_;    // 8192
  const int smBlocks = (B_ * H_ * 32 + TPB - 1) / TPB;

  const float* prevOut = x;
  for (int l = 0; l < D_; ++l) {
    const float* Wrl = Wr + (size_t)l * U_ * NR;
    const float* brl = br + (size_t)l * NR;
    const float* Wwl = Ww + (size_t)l * U_ * NW;
    const float* bwl = bw + (size_t)l * NW;
    const float* Wkl = Wk + (size_t)l * U_ * U_;
    const float* bkl = bk + (size_t)l * U_;

    {  // rlog = memstate @ Wr + br : 513 column tiles, 4 row tiles per wave
      gemm_bias_wmma<0, NR, 0><<<(NR / 16 + 7) / 8, TPB, 0, stream>>>(
          memstate_bf, Wrl, brl, rlog, nullptr);
    }
    softmax_slots<<<smBlocks, TPB, 0, stream>>>(rlog, MP1);
    weighted_read<<<gBU4, TPB, 0, stream>>>(
        (const float4*)mem, rlog, (const float4*)prevOut, (uint2*)rvec_bf);
    {  // newval = relu(r @ Wk + bk), dual f32 + bf16 output
      gemm_bias_wmma<1, U_, 1><<<(U_ / 16 + 7) / 8, TPB, 0, stream>>>(
          rvec_bf, Wkl, bkl, newv, newv_bf);
    }
    {  // wlog = newval @ Ww + bw
      gemm_bias_wmma<0, NW, 0><<<(NW / 16 + 7) / 8, TPB, 0, stream>>>(
          newv_bf, Wwl, bwl, wlog, nullptr);
    }
    softmax_slots<<<smBlocks, TPB, 0, stream>>>(wlog, M_);
    mem_update<<<gBU4, TPB, 0, stream>>>(
        (float4*)mem, wlog, (const float4*)newv, (uint2*)memstate_bf,
        (float4*)outv, (l == D_ - 1) ? 1 : 0);
    prevOut = newv;
  }
}
